// GraphCNN_7636451852422
// MI455X (gfx1250) — matrix operand order repaired and checked by reading, not compiled, not run on hardware
//
#include <hip/hip_runtime.h>

typedef __attribute__((ext_vector_type(16))) __bf16 v16bf;
typedef __attribute__((ext_vector_type(8)))  __bf16 v8bf;
typedef __attribute__((ext_vector_type(8)))  float  v8f;
typedef int v4i __attribute__((vector_size(16)));

#define AS_GLOBAL __attribute__((address_space(1)))
#define AS_LDS    __attribute__((address_space(3)))

#if defined(__gfx1250__) && __has_builtin(__builtin_amdgcn_global_load_async_to_lds_b128)
#define USE_ASYNC_LDS 1
#else
#define USE_ASYNC_LDS 0
#endif

// Shapes
static constexpr int kB = 256;
static constexpr int kN = 128;
static constexpr int kD = 512;
static constexpr int kE = 2048;
static constexpr int kR = 1024;
static constexpr int kO = 10;

__device__ __forceinline__ float fast_tanh(float x) {
#if __has_builtin(__builtin_amdgcn_tanhf)
    return __builtin_amdgcn_tanhf(x);          // v_tanh_f32 (CDNA5 TRANS op)
#elif __has_builtin(__builtin_amdgcn_tanh_f32)
    return __builtin_amdgcn_tanh_f32(x);
#else
    return tanhf(x);
#endif
}

__device__ __forceinline__ void wait_async0() {
#if __has_builtin(__builtin_amdgcn_s_wait_asynccnt)
    __builtin_amdgcn_s_wait_asynccnt(0);
#else
    asm volatile("s_wait_asynccnt 0x0" ::: "memory");
#endif
}

#if USE_ASYNC_LDS
__device__ __forceinline__ void async_load_b128(const void* g, void* l) {
    __builtin_amdgcn_global_load_async_to_lds_b128(
        (AS_GLOBAL v4i*)g, (AS_LDS v4i*)l, 0, 0);
}
#endif

// ---------------------------------------------------------------------------
// Precision staging: h (f32) -> bf16, W (f32, [k][c]) -> bf16 transposed [c][k]
// ---------------------------------------------------------------------------
__global__ void convert_h(const float* __restrict__ src, __bf16* __restrict__ dst, int n4)
{
    int i = blockIdx.x * blockDim.x + threadIdx.x;
    const int stride = gridDim.x * blockDim.x;
    for (; i < n4; i += stride) {
        const float4 v = ((const float4*)src)[i];
        __bf16* d = dst + (size_t)i * 4;
        d[0] = (__bf16)v.x; d[1] = (__bf16)v.y;
        d[2] = (__bf16)v.z; d[3] = (__bf16)v.w;
    }
}

__global__ __launch_bounds__(256)
void convert_w(const float* __restrict__ W, __bf16* __restrict__ Wt)
{
    const int l = blockIdx.y;                  // layer
    const int c = blockIdx.x * 256 + threadIdx.x;
    const float* ws = W + (size_t)l * (kD + 1) * kR + c;   // W[l][k][c]
    __bf16* wd = Wt + ((size_t)l * kR + c) * kD;           // Wt[l][c][k]
    for (int k = 0; k < kD; ++k)
        wd[k] = (__bf16)ws[(size_t)k * kR];
}

// ---------------------------------------------------------------------------
// Fused CP-pool: pool[b,c] = prod_n tanh( h[b,n,:]·W[:,c] + bias[c] )
// 8 waves per WG compute the full 128(N) x 128(cols) tile; async global->LDS
// staging (double-buffered, compile-time buffer select) feeds
// v_wmma_f32_16x16x32_bf16.
// ---------------------------------------------------------------------------
__global__ __launch_bounds__(256)
void cp_pool_wmma(const __bf16* __restrict__ hbf, const __bf16* __restrict__ Wt,
                  const float* __restrict__ bias, float* __restrict__ pool)
{
    __shared__ __attribute__((aligned(64))) __bf16 At[2][128][32];  // [n][k]
    __shared__ __attribute__((aligned(64))) __bf16 Bt[2][128][32];  // [c][k]
    __shared__ float red[8][128];

    const int b    = blockIdx.y;
    const int c0   = blockIdx.x * 128;
    const int t    = threadIdx.x;
    const int lane = t & 31;
    const int wv   = t >> 5;       // wave -> rows 16*wv .. 16*wv+15
    const int kh   = lane >> 4;    // K-half select: lane half kh holds k = 16*kh .. 16*kh+15 for A and B
    const int cl   = lane & 15;

    const __bf16* ha = hbf + (size_t)b * kN * kD;   // A rows, stride kD
    const __bf16* wa = Wt + (size_t)c0 * kD;        // B cols (transposed), stride kD

    v8f acc[8] = {};

#if USE_ASYNC_LDS
    auto issue = [&](int kt, int buf) {
        const int k0 = kt * 32;
        #pragma unroll
        for (int rep = 0; rep < 2; ++rep) {
            const int ch  = t + rep * 256;          // 512 16B chunks per tile
            const int row = ch >> 2;
            const int kq  = ch & 3;
            async_load_b128(ha + (size_t)row * kD + k0 + kq * 8,
                            &At[buf][row][kq * 8]);
            async_load_b128(wa + (size_t)row * kD + k0 + kq * 8,
                            &Bt[buf][row][kq * 8]);
        }
    };
#else
    auto stage_sync = [&](int kt, int buf) {
        const int k0 = kt * 32;
        #pragma unroll
        for (int rep = 0; rep < 2; ++rep) {
            const int ch  = t + rep * 256;
            const int row = ch >> 2;
            const int kq  = ch & 3;
            *(v8bf*)&At[buf][row][kq * 8] =
                *(const v8bf*)(ha + (size_t)row * kD + k0 + kq * 8);
            *(v8bf*)&Bt[buf][row][kq * 8] =
                *(const v8bf*)(wa + (size_t)row * kD + k0 + kq * 8);
        }
    };
#endif

    auto compute = [&](int buf) {
        const int arow = 16 * wv + cl;
        const v16bf a = *(const v16bf*)&At[buf][arow][16 * kh];
        #pragma unroll
        for (int ct = 0; ct < 8; ++ct) {
            const v16bf bm = *(const v16bf*)&Bt[buf][ct * 16 + cl][16 * kh];
            acc[ct] = __builtin_amdgcn_wmma_f32_16x16x32_bf16(
                false, a, false, bm, (short)0, acc[ct], false, false);
        }
    };

#if USE_ASYNC_LDS
    issue(0, 0);
    #pragma unroll 1
    for (int kt = 0; kt < 16; kt += 2) {
        // ---- even step: compute buf0, prefetch kt+1 into buf1 ----
        wait_async0();
        __syncthreads();
        issue(kt + 1, 1);               // kt+1 <= 15, always valid
        compute(0);
        // ---- odd step: compute buf1, prefetch kt+2 into buf0 ----
        wait_async0();
        __syncthreads();
        if (kt < 14) issue(kt + 2, 0);
        compute(1);
    }
#else
    for (int kt = 0; kt < 16; ++kt) {
        __syncthreads();
        stage_sync(kt, 0);
        __syncthreads();
        compute(0);
    }
#endif

    // ---- epilogue: +bias, tanh, product over rows ----
    #pragma unroll
    for (int ct = 0; ct < 8; ++ct) {
        const float bs = bias[c0 + ct * 16 + cl];
        float p = 1.0f;
        #pragma unroll
        for (int v = 0; v < 8; ++v) p *= fast_tanh(acc[ct][v] + bs);
        p *= __shfl_xor(p, 16, 32);                 // combine M and M+8 halves
        if (lane < 16) red[wv][ct * 16 + cl] = p;   // rows 16wv..16wv+15
    }
    __syncthreads();
    if (t < 128) {
        float p = red[0][t];
        #pragma unroll
        for (int w = 1; w < 8; ++w) p *= red[w][t];
        pool[(size_t)b * kR + c0 + t] = p;
    }
}

// ---------------------------------------------------------------------------
// score[b,:] (+)= pool[b,:] @ lin_w + lin_b
// ---------------------------------------------------------------------------
__global__ __launch_bounds__(256)
void score_kernel(const float* __restrict__ pool, const float* __restrict__ lw,
                  const float* __restrict__ lb, float* __restrict__ out, int first)
{
    __shared__ float red[kO][256];
    const int b = blockIdx.x, t = threadIdx.x;
    float part[kO];
    #pragma unroll
    for (int o = 0; o < kO; ++o) part[o] = 0.0f;
    for (int r = t; r < kR; r += 256) {
        const float v = pool[(size_t)b * kR + r];
        const float* wr = lw + (size_t)r * kO;
        #pragma unroll
        for (int o = 0; o < kO; ++o) part[o] += v * wr[o];
    }
    #pragma unroll
    for (int o = 0; o < kO; ++o) red[o][t] = part[o];
    __syncthreads();
    for (int s = 128; s > 0; s >>= 1) {
        if (t < s) {
            #pragma unroll
            for (int o = 0; o < kO; ++o) red[o][t] += red[o][t + s];
        }
        __syncthreads();
    }
    if (t < kO) {
        const float v = red[t][0] + lb[t];
        if (first) out[b * kO + t] = v;
        else       out[b * kO + t] += v;
    }
}

// ---------------------------------------------------------------------------
// Aggregation: h_next = h_cur (copy), then h_next[b,dst] += h_cur[b,src]
// ---------------------------------------------------------------------------
__global__ void copy_f4(const float4* __restrict__ s, float4* __restrict__ d, int n)
{
    int i = blockIdx.x * blockDim.x + threadIdx.x;
    const int stride = gridDim.x * blockDim.x;
    for (; i < n; i += stride) d[i] = s[i];
}

__global__ __launch_bounds__(256)
void scatter_add(const float* __restrict__ hsrc, const int* __restrict__ edges,
                 float* __restrict__ hdst)
{
    const int gid  = blockIdx.x * blockDim.x + threadIdx.x;
    const int wave = gid >> 5;               // one wave per edge
    const int lane = gid & 31;
    const int b = wave >> 11;                // E = 2048
    const int e = wave & (kE - 1);
    const int sn = edges[(size_t)b * 2 * kE + e];
    const int dn = edges[(size_t)b * 2 * kE + kE + e];
    const float* sp = hsrc + ((size_t)b * kN + sn) * kD;
    float*       dp = hdst + ((size_t)b * kN + dn) * kD;
    #pragma unroll 4
    for (int i = lane; i < kD; i += 32)
        atomicAdd(&dp[i], sp[i]);            // global_atomic_add_f32 (L2-resident)
}

// ---------------------------------------------------------------------------
extern "C" void kernel_launch(void* const* d_in, const int* in_sizes, int n_in,
                              void* d_out, int out_size, void* d_ws, size_t ws_size,
                              hipStream_t stream)
{
    (void)in_sizes; (void)n_in; (void)out_size; (void)ws_size;
    const float* x     = (const float*)d_in[0];
    const int*   edges = (const int*)d_in[1];
    const float* W_cp  = (const float*)d_in[2];   // (5, 513, 1024)
    const float* lin_w = (const float*)d_in[3];   // (5, 1024, 10)
    const float* lin_b = (const float*)d_in[4];   // (5, 10)
    float* out = (float*)d_out;                   // (256, 10)

    const size_t hElems = (size_t)kB * kN * kD;   // 16.78M floats (67 MB)
    float*  hA   = (float*)d_ws;
    float*  hB   = hA + hElems;
    float*  pool = hB + hElems;                   // (256, 1024)
    __bf16* hbf  = (__bf16*)(pool + (size_t)kB * kR);  // 33.5 MB
    __bf16* Wt   = hbf + hElems;                  // (5, 1024, 512) bf16

    convert_w<<<dim3(kR / 256, 5), 256, 0, stream>>>(W_cp, Wt);

    const float* hcur = x;
    for (int l = 0; l < 5; ++l) {
        if (l > 0) {
            float* hn = (l & 1) ? hA : hB;        // ping-pong, never read==write
            copy_f4<<<4096, 256, 0, stream>>>((const float4*)hcur, (float4*)hn,
                                              (int)(hElems / 4));
            scatter_add<<<(kB * kE) / 8, 256, 0, stream>>>(hcur, edges, hn);
            hcur = hn;
        }
        convert_h<<<4096, 256, 0, stream>>>(hcur, hbf, (int)(hElems / 4));
        cp_pool_wmma<<<dim3(kR / 128, kB), 256, 0, stream>>>(
            hbf, Wt + (size_t)l * kR * kD,
            W_cp + (size_t)l * (kD + 1) * kR + (size_t)kD * kR, pool);
        score_kernel<<<kB, 256, 0, stream>>>(
            pool, lin_w + (size_t)l * kR * kO, lin_b + l * kO, out, l == 0);
    }
}
